// GATClassifier_7447473291954
// MI455X (gfx1250) — compile-verified, matrix operand order checked by reading, not run on hardware
//
#include <hip/hip_runtime.h>
#include <hip/hip_bf16.h>

typedef __bf16 v16bf __attribute__((ext_vector_type(16)));
typedef float  v8f   __attribute__((ext_vector_type(8)));

#define HEADS 4
#define NEG_SLOPE 0.2f

// ---------------- generic fill ----------------
__global__ void fill_f32(float* __restrict__ p, float v, long long n) {
  long long i = (long long)blockIdx.x * blockDim.x + threadIdx.x;
  if (i < n) p[i] = v;
}

// ---------------- GEMM via bf16 WMMA, fp32 accumulate, LDS-staged B ----------------
// Out[nrows x MOUT] = op(A)[nrows x KDIM] @ W[KDIM x MOUT]
// op(A) = relu(A + bias) when RELU_BIAS, else A.
// W is staged once per block into LDS, bf16-converted and pre-swizzled into the
// WMMA B-fragment lane layout. Each wave then computes a 16-row x MOUT slab:
// A fragment loaded once per K-step (b128 x4), B fragments read from LDS (32B),
// MOUT/16 back-to-back WMMAs per K-step. nrows%16==0, KDIM%32==0, MOUT%16==0.
template <int MOUT, int KDIM, bool RELU_BIAS>
__global__ void gemm_wmma_bf16(const float* __restrict__ A,
                               const float* __restrict__ W,
                               const float* __restrict__ bias,
                               float* __restrict__ Out,
                               int nrows) {
  constexpr int NT    = MOUT / 16;
  constexpr int KSTEP = KDIM / 32;
  constexpr int TOTAL = KDIM * MOUT;
  __shared__ __align__(32) __bf16 Bf[TOTAL];   // fragment layout [kstep][tile][lane][16]

  const int lane = threadIdx.x & 31;
  const int wave = threadIdx.x >> 5;
  const int half = lane >> 4;                  // 0: lanes 0-15, 1: lanes 16-31
  const int l16  = lane & 15;
  const int row0 = (blockIdx.x * (blockDim.x >> 5) + wave) * 16;

  // ---- cooperative staging: coalesced read of W, swizzled bf16 store to LDS ----
  for (int idx = threadIdx.x; idx < TOTAL; idx += blockDim.x) {
    int k = idx / MOUT, col = idx % MOUT;      // consecutive tids -> consecutive cols
    int r = k & 31, kb = k >> 5;
    int hf, i;
    if      (r <  8) { hf = 0; i = r;      }   // elems 0..7  = K kc..kc+7
    else if (r < 16) { hf = 1; i = r - 8;  }
    else if (r < 24) { hf = 0; i = r - 8;  }   // elems 8..15 = K kc+16..kc+23
    else             { hf = 1; i = r - 16; }
    int l = hf * 16 + (col & 15);
    int t = col >> 4;
    Bf[((kb * NT + t) * 32 + l) * 16 + i] = (__bf16)W[(size_t)k * MOUT + col];
  }
  __syncthreads();

  if (row0 < nrows) {                          // wave-uniform (EXEC all-1s at WMMA)
    v8f acc[NT] = {};
    #pragma unroll
    for (int kb = 0; kb < KSTEP; ++kb) {
      const int kc = kb * 32 + half * 8;       // this lane's K chunk base
      // A fragment (ISA 16-bit A 16x32 layout), vectorized b128 loads
      const float* ap = A + (size_t)(row0 + l16) * KDIM + kc;
      float va[16];
      *(float4*)(va + 0)  = *(const float4*)(ap + 0);
      *(float4*)(va + 4)  = *(const float4*)(ap + 4);
      *(float4*)(va + 8)  = *(const float4*)(ap + 16);
      *(float4*)(va + 12) = *(const float4*)(ap + 20);
      if (RELU_BIAS) {
        const float* bp = bias + kc;
        float vb[16];
        *(float4*)(vb + 0)  = *(const float4*)(bp + 0);
        *(float4*)(vb + 4)  = *(const float4*)(bp + 4);
        *(float4*)(vb + 8)  = *(const float4*)(bp + 16);
        *(float4*)(vb + 12) = *(const float4*)(bp + 20);
        #pragma unroll
        for (int i = 0; i < 16; ++i) {
          float v = va[i] + vb[i];
          va[i] = v > 0.f ? v : 0.f;
        }
      }
      v16bf a;
      #pragma unroll
      for (int i = 0; i < 16; ++i) a[i] = (__bf16)va[i];

      #pragma unroll
      for (int t = 0; t < NT; ++t) {
        v16bf b = *(const v16bf*)&Bf[((kb * NT + t) * 32 + lane) * 16];
        acc[t] = __builtin_amdgcn_wmma_f32_16x16x32_bf16(
            false, a, false, b, (short)0, acc[t], false, false);
      }
    }
    // C/D layout: VGPR r -> row (half*8 + r), col = l16
    #pragma unroll
    for (int t = 0; t < NT; ++t) {
      float* op = Out + (size_t)(row0 + half * 8) * MOUT + t * 16 + l16;
      #pragma unroll
      for (int r = 0; r < 8; ++r) op[(size_t)r * MOUT] = acc[t][r];
    }
  }
}

// ---------------- per-node attention logits (wave per node, coalesced) ----------------
template <int C>
__global__ void attn_logits(const float* __restrict__ Hm,
                            const float* __restrict__ a_s,
                            const float* __restrict__ a_d,
                            float* __restrict__ es, float* __restrict__ ed, int n) {
  const int lane = threadIdx.x & 31;
  const int node = (int)(((long long)blockIdx.x * blockDim.x + threadIdx.x) >> 5);
  if (node >= n) return;
  const float* hp = Hm + (size_t)node * (HEADS * C);
  if (C == 32) {                                 // one head per 32 lanes
    #pragma unroll
    for (int h = 0; h < HEADS; ++h) {
      float v = hp[h * 32 + lane];
      float s = v * a_s[h * 32 + lane];
      float d = v * a_d[h * 32 + lane];
      #pragma unroll
      for (int m = 16; m; m >>= 1) { s += __shfl_xor(s, m, 32); d += __shfl_xor(d, m, 32); }
      if (lane == 0) { es[node * HEADS + h] = s; ed[node * HEADS + h] = d; }
    }
  } else {                                       // C==16: two heads per 32 lanes
    #pragma unroll
    for (int i = 0; i < 2; ++i) {
      int h = i * 2 + (lane >> 4);
      int c = lane & 15;
      float v = hp[h * 16 + c];
      float s = v * a_s[h * 16 + c];
      float d = v * a_d[h * 16 + c];
      #pragma unroll
      for (int m = 8; m; m >>= 1) { s += __shfl_xor(s, m, 32); d += __shfl_xor(d, m, 32); }
      if (c == 0) { es[node * HEADS + h] = s; ed[node * HEADS + h] = d; }
    }
  }
}

// ---------------- edge helpers ----------------
__device__ inline void edge_endpoints(const long long* __restrict__ ei,
                                      int e, int E, int& src, int& dst) {
  if (e < E) { src = (int)ei[e]; dst = (int)ei[(size_t)E + e]; }
  else       { src = dst = e - E; }              // appended self-loops
}

// pass 1: e = leaky_relu(es[src]+ed[dst]); segment-max into mx[dst]
__global__ void edge_logits_max(const long long* __restrict__ ei,
                                const float* __restrict__ es,
                                const float* __restrict__ ed,
                                float* __restrict__ ee, float* __restrict__ mx,
                                int Etot, int E) {
  int e = blockIdx.x * blockDim.x + threadIdx.x;
  if (e >= Etot) return;
  int s, d; edge_endpoints(ei, e, E, s, d);
  #pragma unroll
  for (int h = 0; h < HEADS; ++h) {
    float v = es[s * HEADS + h] + ed[d * HEADS + h];
    v = v > 0.f ? v : NEG_SLOPE * v;
    ee[(size_t)e * HEADS + h] = v;
    atomicMax(&mx[d * HEADS + h], v);            // global_atomic_max_num_f32
  }
}

// pass 2: ex = exp(e - mx[dst]); segment-sum into sm[dst]; ee overwritten with ex
__global__ void edge_exp_sum(const long long* __restrict__ ei,
                             float* __restrict__ ee,
                             const float* __restrict__ mx,
                             float* __restrict__ sm, int Etot, int E) {
  int e = blockIdx.x * blockDim.x + threadIdx.x;
  if (e >= Etot) return;
  int s, d; edge_endpoints(ei, e, E, s, d);
  #pragma unroll
  for (int h = 0; h < HEADS; ++h) {
    float x = __expf(ee[(size_t)e * HEADS + h] - mx[d * HEADS + h]);
    ee[(size_t)e * HEADS + h] = x;
    atomicAdd(&sm[d * HEADS + h], x);
  }
}

// pass 3: agg[dst,h,c] += H[src,h,c] * alpha[h]. One wave per edge; alpha hoisted.
template <int C>
__global__ void edge_scatter(const long long* __restrict__ ei,
                             const float* __restrict__ ee,
                             const float* __restrict__ sm,
                             const float* __restrict__ Hm,
                             float* __restrict__ agg, int Etot, int E) {
  const int wid  = (int)(((long long)blockIdx.x * blockDim.x + threadIdx.x) >> 5);
  const int lane = threadIdx.x & 31;
  if (wid >= Etot) return;
  int s, d; edge_endpoints(ei, wid, E, s, d);
  float alpha[HEADS];
  #pragma unroll
  for (int h = 0; h < HEADS; ++h)
    alpha[h] = ee[(size_t)wid * HEADS + h] / (sm[d * HEADS + h] + 1e-16f);
  constexpr int HC = HEADS * C;
  #pragma unroll
  for (int c0 = 0; c0 < HC; c0 += 32) {
    int c = c0 + lane;
    int h = c / C;                               // compile-time shift
    atomicAdd(&agg[(size_t)d * HC + c], Hm[(size_t)s * HC + c] * alpha[h]);
  }
}

// ---------------- global add pool (64 channels) ----------------
__global__ void pool_sum(const float* __restrict__ agg2,
                         float* __restrict__ pooled, int n) {
  __shared__ float acc[64];
  int tid = threadIdx.x;                         // 256 threads
  int c = tid & 63, g = tid >> 6;                // 4 row groups x 64 channels
  if (tid < 64) acc[tid] = 0.f;
  __syncthreads();
  int base = blockIdx.x * 64;
  float loc = 0.f;
  for (int r = g; r < 64; r += 4) {
    int nn = base + r;
    if (nn < n) loc += agg2[(size_t)nn * 64 + c];
  }
  atomicAdd(&acc[c], loc);
  __syncthreads();
  if (tid < 64) atomicAdd(&pooled[tid], acc[tid]);
}

// ---------------- MLP head ----------------
__global__ void head_mlp(const float* __restrict__ pooled,
                         const float* __restrict__ b2,
                         const float* __restrict__ fcw, const float* __restrict__ fcb,
                         const float* __restrict__ fc2w, const float* __restrict__ fc2b,
                         float* __restrict__ out, int n) {
  __shared__ float g1[32];
  int j = threadIdx.x;                           // 32 threads
  float a = fcb[j];
  for (int c = 0; c < 64; ++c)
    a += (pooled[c] + (float)n * b2[c]) * fcw[c * 32 + j];
  g1[j] = a;
  __syncthreads();
  if (j < 2) {
    float o = fc2b[j];
    for (int k = 0; k < 32; ++k) o += g1[k] * fc2w[k * 2 + j];
    out[j] = o;
  }
}

extern "C" void kernel_launch(void* const* d_in, const int* in_sizes, int n_in,
                              void* d_out, int out_size, void* d_ws, size_t ws_size,
                              hipStream_t stream) {
  const float*     x    = (const float*)d_in[0];
  const long long* ei   = (const long long*)d_in[1];
  const float*     W1   = (const float*)d_in[2];
  const float*     a1s  = (const float*)d_in[3];
  const float*     a1d  = (const float*)d_in[4];
  const float*     b1   = (const float*)d_in[5];
  const float*     W2   = (const float*)d_in[6];
  const float*     a2s  = (const float*)d_in[7];
  const float*     a2d  = (const float*)d_in[8];
  const float*     b2   = (const float*)d_in[9];
  const float*     fcw  = (const float*)d_in[10];
  const float*     fcb  = (const float*)d_in[11];
  const float*     fc2w = (const float*)d_in[12];
  const float*     fc2b = (const float*)d_in[13];

  const int FIN = 128, HC1 = 128, HC2 = 64;
  const int N    = in_sizes[0] / FIN;
  const int E    = in_sizes[1] / 2;
  const int Etot = E + N;

  float* ws = (float*)d_ws;
  size_t off = 0;
  float* h1     = ws + off; off += (size_t)N * HC1;      // h1, later reused as h2
  float* agg1   = ws + off; off += (size_t)N * HC1;      // agg1, later reused as agg2
  float* ee     = ws + off; off += (size_t)Etot * HEADS;
  float* es     = ws + off; off += (size_t)N * HEADS;
  float* ed     = ws + off; off += (size_t)N * HEADS;
  float* mx     = ws + off; off += (size_t)N * HEADS;
  float* sm     = ws + off; off += (size_t)N * HEADS;
  float* pooled = ws + off; off += 64;
  (void)ws_size; (void)n_in; (void)out_size;
  float* h2   = h1;                                      // buffer reuse (h1 dead by then)
  float* agg2 = agg1;                                    // buffer reuse (agg1 dead by then)

  auto blks = [](long long n, int b) { return (unsigned)((n + b - 1) / b); };
  const float NEG_INF = -__builtin_inff();
  const unsigned gemmBlocks = blks((long long)(N / 16), 8);   // 8 waves/block

  // ---- layer 1: h1 = x @ W1 (bf16 WMMA, LDS-staged B) ----
  gemm_wmma_bf16<128, 128, false><<<gemmBlocks, 256, 0, stream>>>(x, W1, nullptr, h1, N);
  attn_logits<32><<<blks((long long)N * 32, 256), 256, 0, stream>>>(h1, a1s, a1d, es, ed, N);
  fill_f32<<<blks((long long)N * HEADS, 256), 256, 0, stream>>>(mx, NEG_INF, (long long)N * HEADS);
  fill_f32<<<blks((long long)N * HEADS, 256), 256, 0, stream>>>(sm, 0.f, (long long)N * HEADS);
  fill_f32<<<blks((long long)N * HC1, 256), 256, 0, stream>>>(agg1, 0.f, (long long)N * HC1);
  edge_logits_max<<<blks(Etot, 256), 256, 0, stream>>>(ei, es, ed, ee, mx, Etot, E);
  edge_exp_sum<<<blks(Etot, 256), 256, 0, stream>>>(ei, ee, mx, sm, Etot, E);
  edge_scatter<32><<<blks((long long)Etot * 32, 256), 256, 0, stream>>>(ei, ee, sm, h1, agg1, Etot, E);

  // ---- layer 2: h2 = relu(agg1 + b1) @ W2 (bias+ReLU fused into A-fragment load) ----
  gemm_wmma_bf16<64, 128, true><<<gemmBlocks, 256, 0, stream>>>(agg1, W2, b1, h2, N);
  attn_logits<16><<<blks((long long)N * 32, 256), 256, 0, stream>>>(h2, a2s, a2d, es, ed, N);
  fill_f32<<<blks((long long)N * HEADS, 256), 256, 0, stream>>>(mx, NEG_INF, (long long)N * HEADS);
  fill_f32<<<blks((long long)N * HEADS, 256), 256, 0, stream>>>(sm, 0.f, (long long)N * HEADS);
  fill_f32<<<blks((long long)N * HC2, 256), 256, 0, stream>>>(agg2, 0.f, (long long)N * HC2);
  edge_logits_max<<<blks(Etot, 256), 256, 0, stream>>>(ei, es, ed, ee, mx, Etot, E);
  edge_exp_sum<<<blks(Etot, 256), 256, 0, stream>>>(ei, ee, mx, sm, Etot, E);
  edge_scatter<16><<<blks((long long)Etot * 32, 256), 256, 0, stream>>>(ei, ee, sm, h2, agg2, Etot, E);

  // ---- pool + head ----
  fill_f32<<<1, 64, 0, stream>>>(pooled, 0.f, 64);
  pool_sum<<<blks(N, 64), 256, 0, stream>>>(agg2, pooled, N);
  head_mlp<<<1, 32, 0, stream>>>(pooled, b2, fcw, fcb, fc2w, fc2b, (float*)d_out, N);
}
